// LocalPNConvNet_90185723281820
// MI455X (gfx1250) — matrix operand order repaired and checked by reading, not compiled, not run on hardware
//
#include <hip/hip_runtime.h>

// ---------------------------------------------------------------------------
// LocalPNConvNet for MI455X (gfx1250): KNN + gather + 9x (bf16 WMMA GEMM with
// TDM (tensor_load_to_lds) double-buffered tile staging and fused BN-stat
// accumulation) + BN-apply/ReLU + max-over-K pooling + concat.
// ---------------------------------------------------------------------------

typedef __bf16 bf16_t;
typedef __attribute__((ext_vector_type(16))) __bf16 v16bf;
typedef __attribute__((ext_vector_type(8)))  __bf16 v8bf;
typedef __attribute__((ext_vector_type(8)))  float  v8f;

typedef __attribute__((ext_vector_type(4))) unsigned int u32x4;
typedef __attribute__((ext_vector_type(8))) int          i32x8;
typedef __attribute__((ext_vector_type(4))) int          i32x4;

typedef __attribute__((address_space(3))) char lds_char_t;

#define NPTS  2048
#define BATCH 8
#define KNBR  16
#define M1ROWS (BATCH * NPTS * KNBR)   // 262144
#define M2ROWS (BATCH * NPTS)          // 16384

__device__ inline unsigned lds_addr_of(void* p) {
  return (unsigned)(uintptr_t)(lds_char_t*)p;
}

// Issue one TDM 2D tile load: rows x 32 bf16 elements, global row stride
// `kstride` elements, packed into LDS with +4-DWORD padding per 16-DWORD row
// (i.e. LDS row stride = 40 bf16 = 80B, matching the [rows][40] arrays).
__device__ inline void tdm_load_tile(unsigned lds_off, const bf16_t* gptr,
                                     int rows, int kstride) {
  unsigned long long ga = (unsigned long long)(uintptr_t)gptr;
  u32x4 g0;
  g0.x = 1u;                                  // count=1 valid descriptor
  g0.y = lds_off;                             // lds_addr
  g0.z = (unsigned)ga;                        // global_addr[31:0]
  g0.w = (unsigned)(ga >> 32) | (2u << 30);   // global_addr[56:32] | type=2
  i32x8 g1;
  g1[0] = (1 << 16)      // data_size = 1 (2 bytes)
        | (1 << 20)      // pad_enable
        | (3 << 22)      // pad_interval: every 16 DWORDs (64B row)
        | (3 << 25);     // pad_amount: 4 DWORDs (16B) -> 80B LDS row stride
  g1[1] = (32 << 16);    // tensor_dim0 = 32 (low 16 bits)
  g1[2] = (rows << 16);  // tensor_dim0 hi=0 | tensor_dim1 low16 = rows
  g1[3] = (32 << 16);    // tensor_dim1 hi=0 | tile_dim0 = 32
  g1[4] = rows;          // tile_dim1 = rows, tile_dim2 = 0 (2D)
  g1[5] = kstride;       // tensor_dim0_stride low32 (elements)
  g1[6] = 0;             // stride hi | tensor_dim1_stride lo (unused, 2D)
  g1[7] = 0;
  i32x4 z4 = {0, 0, 0, 0};
#if defined(__clang_major__) && (__clang_major__ >= 23)
  i32x8 z8 = {0, 0, 0, 0, 0, 0, 0, 0};
  __builtin_amdgcn_tensor_load_to_lds(g0, g1, z4, z4, z8, 0);
#else
  __builtin_amdgcn_tensor_load_to_lds(g0, g1, z4, z4, 0);
#endif
}

// ---------------------------------------------------------------------------
__global__ void zero_f32_kernel(float* __restrict__ p, int n) {
  int i = blockIdx.x * blockDim.x + threadIdx.x;
  if (i < n) p[i] = 0.f;
}

// ---------------------------------------------------------------------------
// KNN: one block = (batch b, 256 query points). All 2048 positions of batch b
// (24 KB) staged in LDS once; each thread keeps a sorted top-16 in registers.
__global__ __launch_bounds__(256)
void knn_kernel(const float* __restrict__ pos, int* __restrict__ nn) {
  __shared__ float sp[NPTS * 3];
  const int b = blockIdx.y;
  for (int t = threadIdx.x; t < NPTS * 3; t += blockDim.x)
    sp[t] = pos[(size_t)b * NPTS * 3 + t];
  __syncthreads();

  const int n = blockIdx.x * blockDim.x + threadIdx.x;
  const float qx = sp[n * 3 + 0], qy = sp[n * 3 + 1], qz = sp[n * 3 + 2];

  float bd[KNBR];
  int   bi[KNBR];
#pragma unroll
  for (int k = 0; k < KNBR; ++k) { bd[k] = 3.4e38f; bi[k] = 0; }

  for (int j = 0; j < NPTS; ++j) {
    float dx = qx - sp[j * 3 + 0];
    float dy = qy - sp[j * 3 + 1];
    float dz = qz - sp[j * 3 + 2];
    float d2 = dx * dx + dy * dy + dz * dz;
    if (d2 < bd[KNBR - 1]) {
      bd[KNBR - 1] = d2; bi[KNBR - 1] = j;
#pragma unroll
      for (int p = KNBR - 1; p > 0; --p) {
        if (bd[p] < bd[p - 1]) {
          float tf = bd[p]; bd[p] = bd[p - 1]; bd[p - 1] = tf;
          int   ti = bi[p]; bi[p] = bi[p - 1]; bi[p - 1] = ti;
        }
      }
    }
  }
  int* dst = nn + ((size_t)b * NPTS + n) * KNBR;
#pragma unroll
  for (int k = 0; k < KNBR; ++k) dst[k] = bi[k];
}

// ---------------------------------------------------------------------------
// Gather neighbors: A0[row, 0:64]=x[b,j,:], [64:67]=pos[b,j,:], [67:96]=0.
__global__ void gather_kernel(const float* __restrict__ x, const float* __restrict__ pos,
                              const int* __restrict__ nn, bf16_t* __restrict__ A0) {
  int row = blockIdx.x * blockDim.x + threadIdx.x;
  if (row >= M1ROWS) return;
  int b = row / (NPTS * KNBR);
  int j = nn[row];
  const float* xs = x + ((size_t)b * NPTS + j) * 64;
  const float* ps = pos + ((size_t)b * NPTS + j) * 3;
  bf16_t* dst = A0 + (size_t)row * 96;
#pragma unroll
  for (int c = 0; c < 64; ++c) dst[c] = (bf16_t)xs[c];
  dst[64] = (bf16_t)ps[0];
  dst[65] = (bf16_t)ps[1];
  dst[66] = (bf16_t)ps[2];
#pragma unroll
  for (int c = 67; c < 96; ++c) dst[c] = (bf16_t)0.f;
}

// ---------------------------------------------------------------------------
__global__ void prep_weight_kernel(const float* __restrict__ W, bf16_t* __restrict__ Wb,
                                   int kin, int kpad, int nout) {
  int i = blockIdx.x * blockDim.x + threadIdx.x;
  if (i >= kpad * nout) return;
  int o = i / kpad, c = i % kpad;
  Wb[i] = (c < kin) ? (bf16_t)W[o * kin + c] : (bf16_t)0.f;
}

// ---------------------------------------------------------------------------
// bf16 GEMM: H[M,Nn] = A[M,Kk] * Wt[Nn,Kk]^T + bias, f32 accumulate via
// v_wmma_f32_16x16x32_bf16. Block tile 128x64, 8 waves (4x2), wave tile 32x32.
// Tiles staged by the Tensor Data Mover (double-buffered, TENSORcnt-tracked,
// overlapped with WMMA compute). Epilogue fuses BN statistic accumulation.
// Requires: M % 128 == 0, Nn % 64 == 0, Kk % 32 == 0 (true for all layers).
__global__ __launch_bounds__(256)
void gemm_bf16_bn_kernel(const bf16_t* __restrict__ A, const bf16_t* __restrict__ Wt,
                         const float* __restrict__ bias, bf16_t* __restrict__ H,
                         float* __restrict__ stats, int M, int Nn, int Kk) {
  __shared__ bf16_t As[2][128][40];   // 32 cols + 8 pad (matches TDM pad cfg)
  __shared__ bf16_t Bs[2][64][40];
  __shared__ float  ssum[64];
  __shared__ float  ssq[64];

  const int t    = threadIdx.x;
  const int lane = t & 31;
  const int wv   = t >> 5;
  const int wm   = wv & 3;         // wave M index (0..3)
  const int wn   = wv >> 2;        // wave N index (0..1)
  const int hl   = lane >> 4;      // lane half (0/1)
  const int l16  = lane & 15;

  const int m0 = blockIdx.x * 128;
  const int n0 = blockIdx.y * 64;

  if (t < 64) { ssum[t] = 0.f; ssq[t] = 0.f; }

  const unsigned ldsA0 = lds_addr_of(&As[0][0][0]);
  const unsigned ldsA1 = lds_addr_of(&As[1][0][0]);
  const unsigned ldsB0 = lds_addr_of(&Bs[0][0][0]);
  const unsigned ldsB1 = lds_addr_of(&Bs[1][0][0]);

  const int nt = Kk >> 5;

  // Prologue: DMA tile 0 into buffer 0 (one issue per block, via wave 0).
  if (wv == 0) {
    tdm_load_tile(ldsA0, A + (size_t)m0 * Kk, 128, Kk);
    tdm_load_tile(ldsB0, Wt + (size_t)n0 * Kk, 64, Kk);
  }

  v8f acc[2][2] = {};

  for (int it = 0; it < nt; ++it) {
    const int cur = it & 1;
    if (wv == 0) {
      if (it + 1 < nt) {
        // Prefetch next K-tile into the other buffer (its readers finished at
        // the end-of-iteration barrier of it-1), then wait for the current
        // tile's 2 descriptors (TENSORcnt in-order per wave).
        const int k1 = (it + 1) << 5;
        if (cur) {
          tdm_load_tile(ldsA0, A + (size_t)m0 * Kk + k1, 128, Kk);
          tdm_load_tile(ldsB0, Wt + (size_t)n0 * Kk + k1, 64, Kk);
        } else {
          tdm_load_tile(ldsA1, A + (size_t)m0 * Kk + k1, 128, Kk);
          tdm_load_tile(ldsB1, Wt + (size_t)n0 * Kk + k1, 64, Kk);
        }
        __builtin_amdgcn_s_wait_tensorcnt(2);
      } else {
        __builtin_amdgcn_s_wait_tensorcnt(0);
      }
    }
    __syncthreads();   // publish current tile to all waves

    // A fragment (ISA 16-bit A 16x32): lane {hl,l16} -> elems 0..7 = K[hl*8..],
    // elems 8..15 = K[16+hl*8..].
    v16bf af[2];
#pragma unroll
    for (int mt = 0; mt < 2; ++mt) {
      int m = wm * 32 + mt * 16 + l16;
      v8bf lo = *(const v8bf*)&As[cur][m][hl * 8];
      v8bf hi = *(const v8bf*)&As[cur][m][16 + hl * 8];
      af[mt] = __builtin_shufflevector(lo, hi, 0, 1, 2, 3, 4, 5, 6, 7,
                                       8, 9, 10, 11, 12, 13, 14, 15);
    }
    // B fragment (32x16, K x N): lane column n = l16, elems 0..7 = K[hl*8..], elems 8..15 = K[16+hl*8..].
    // Wt is [n][k] == B^T, so the access pattern mirrors A.
    v16bf bfr[2];
#pragma unroll
    for (int nt2 = 0; nt2 < 2; ++nt2) {
      int n = wn * 32 + nt2 * 16 + l16;
      v8bf lo = *(const v8bf*)&Bs[cur][n][hl * 8];
      v8bf hi = *(const v8bf*)&Bs[cur][n][16 + hl * 8];
      bfr[nt2] = __builtin_shufflevector(lo, hi, 0, 1, 2, 3, 4, 5, 6, 7,
                                         8, 9, 10, 11, 12, 13, 14, 15);
    }

#pragma unroll
    for (int mt = 0; mt < 2; ++mt)
#pragma unroll
      for (int nt2 = 0; nt2 < 2; ++nt2)
        acc[mt][nt2] = __builtin_amdgcn_wmma_f32_16x16x32_bf16(
            false, af[mt], false, bfr[nt2], (short)0, acc[mt][nt2], false, false);

    __syncthreads();   // all reads of this buffer done -> safe to DMA over it
  }

  // Epilogue: bias, bf16 store, fused BN statistics.
#pragma unroll
  for (int nt2 = 0; nt2 < 2; ++nt2) {
    const int jn  = wn * 32 + nt2 * 16 + l16;   // block-local channel
    const int col = n0 + jn;
    const float bv = bias[col];
    float s1 = 0.f, s2 = 0.f;
#pragma unroll
    for (int mt = 0; mt < 2; ++mt) {
#pragma unroll
      for (int r = 0; r < 8; ++r) {
        int row = m0 + wm * 32 + mt * 16 + hl * 8 + r;  // D: M = r + 8*half
        float v = acc[mt][nt2][r] + bv;
        H[(size_t)row * Nn + col] = (bf16_t)v;
        s1 += v;
        s2 += v * v;
      }
    }
    atomicAdd(&ssum[jn], s1);
    atomicAdd(&ssq[jn], s2);
  }
  __syncthreads();
  if (t < 64) {
    atomicAdd(&stats[n0 + t], ssum[t]);
    atomicAdd(&stats[Nn + n0 + t], ssq[t]);
  }
}

// ---------------------------------------------------------------------------
__global__ void finalize_stats_kernel(const float* __restrict__ stats,
                                      const float* __restrict__ gamma,
                                      const float* __restrict__ beta,
                                      float* __restrict__ coef, int Nn, float invM) {
  int c = threadIdx.x;
  if (c < Nn) {
    float mean = stats[c] * invM;
    float var  = stats[Nn + c] * invM - mean * mean;
    float a = gamma[c] * rsqrtf(var + 1e-5f);
    coef[c]       = a;
    coef[256 + c] = beta[c] - mean * a;
  }
}

__global__ void bn_act_kernel(bf16_t* __restrict__ H, const float* __restrict__ coef,
                              int count, int Nn, int relu) {
  int i = blockIdx.x * blockDim.x + threadIdx.x;
  if (i >= count) return;
  int c = i % Nn;
  float v = (float)H[i] * coef[c] + coef[256 + c];
  if (relu) v = fmaxf(v, 0.f);
  H[i] = (bf16_t)v;
}

__global__ void bn_final_kernel(const bf16_t* __restrict__ H, const float* __restrict__ coef,
                                float* __restrict__ out, int count, int Nn) {
  int i = blockIdx.x * blockDim.x + threadIdx.x;
  if (i >= count) return;
  int c = i % Nn;
  out[i] = (float)H[i] * coef[c] + coef[256 + c];
}

// ---------------------------------------------------------------------------
__global__ void maxk_kernel(const bf16_t* __restrict__ in, bf16_t* __restrict__ out,
                            int m2, int Nn) {
  int i = blockIdx.x * blockDim.x + threadIdx.x;
  if (i >= m2 * Nn) return;
  int m = i / Nn, c = i % Nn;
  float best = -3.4e38f;
#pragma unroll
  for (int k = 0; k < KNBR; ++k)
    best = fmaxf(best, (float)in[((size_t)m * KNBR + k) * Nn + c]);
  out[i] = (bf16_t)best;
}

// comb input: [h(128) | broadcast glb(128)] per neighbor row.
__global__ void concat_kernel(const bf16_t* __restrict__ h, const bf16_t* __restrict__ g,
                              bf16_t* __restrict__ out) {
  int i = blockIdx.x * blockDim.x + threadIdx.x;
  if (i >= M1ROWS * 256) return;
  int row = i >> 8, c = i & 255;
  out[i] = (c < 128) ? h[(size_t)row * 128 + c]
                     : g[(size_t)(row >> 4) * 128 + (c - 128)];
}

// ---------------------------------------------------------------------------
extern "C" void kernel_launch(void* const* d_in, const int* in_sizes, int n_in,
                              void* d_out, int out_size, void* d_ws, size_t ws_size,
                              hipStream_t stream) {
  (void)in_sizes; (void)n_in; (void)out_size; (void)ws_size;
  const float* x   = (const float*)d_in[0];
  const float* pos = (const float*)d_in[1];
  float* out = (float*)d_out;

  char* ws = (char*)d_ws;
  size_t off = 0;
  auto arena = [&](size_t bytes) -> void* {
    void* p = ws + off;
    off += (bytes + 255) & ~(size_t)255;
    return p;
  };

  int*    nn    = (int*)   arena((size_t)M2ROWS * KNBR * 4);
  bf16_t* P0    = (bf16_t*)arena((size_t)M1ROWS * 256 * 2);  // A0 (stride 96) -> comb_in (stride 256)
  bf16_t* P1    = (bf16_t*)arena((size_t)M1ROWS * 128 * 2);
  bf16_t* P2    = (bf16_t*)arena((size_t)M1ROWS * 128 * 2);
  bf16_t* P3    = (bf16_t*)arena((size_t)M1ROWS * 128 * 2);
  bf16_t* tmax  = (bf16_t*)arena((size_t)M2ROWS * 128 * 2);
  bf16_t* tg0   = (bf16_t*)arena((size_t)M2ROWS * 256 * 2);
  bf16_t* tg1   = (bf16_t*)arena((size_t)M2ROWS * 128 * 2);
  bf16_t* tcmax = (bf16_t*)arena((size_t)M2ROWS * 128 * 2);
  bf16_t* td0   = (bf16_t*)arena((size_t)M2ROWS * 256 * 2);
  bf16_t* td1   = (bf16_t*)arena((size_t)M2ROWS * 128 * 2);
  float*  stats = (float*) arena((size_t)9 * 512 * 4);
  float*  coef  = (float*) arena((size_t)9 * 512 * 4);

  // Layers: L0,L1,L2 (local), G0,G1 (glb), C0,C1 (comb), D0,D1 (cglb)
  const int kin [9] = { 67,  64, 128, 128, 256, 256, 128, 128, 256};
  const int kpad[9] = { 96,  64, 128, 128, 256, 256, 128, 128, 256};
  const int nout[9] = { 64, 128, 128, 256, 128, 128, 128, 256, 128};
  const int win [9] = {  2,   6,  10,  14,  18,  22,  26,  30,  34};
  const int mrow[9] = {M1ROWS, M1ROWS, M1ROWS, M2ROWS, M2ROWS,
                       M1ROWS, M1ROWS, M2ROWS, M2ROWS};
  const int relu[9] = {1, 1, 0, 1, 0, 1, 0, 1, 0};
  bf16_t* Ain [9] = {P0, P1, P2, tmax, tg0, P0, P1, tcmax, td0};
  bf16_t* Hout[9] = {P1, P2, P3, tg0,  tg1, P1, P2, td0,   td1};

  bf16_t* wbuf[9];
  for (int i = 0; i < 9; ++i)
    wbuf[i] = (bf16_t*)arena((size_t)nout[i] * kpad[i] * 2);

  // 0) zero BN statistic accumulators (deterministic per launch)
  zero_f32_kernel<<<(9 * 512 + 255) / 256, 256, 0, stream>>>(stats, 9 * 512);

  // 1) KNN + gather
  knn_kernel<<<dim3(NPTS / 256, BATCH), 256, 0, stream>>>(pos, nn);
  gather_kernel<<<(M1ROWS + 255) / 256, 256, 0, stream>>>(x, pos, nn, P0);

  // 2) weight conversion (f32 -> padded bf16)
  for (int i = 0; i < 9; ++i) {
    int cnt = nout[i] * kpad[i];
    prep_weight_kernel<<<(cnt + 255) / 256, 256, 0, stream>>>(
        (const float*)d_in[win[i]], wbuf[i], kin[i], kpad[i], nout[i]);
  }

  auto run_layer = [&](int i) {
    dim3 g(mrow[i] / 128, nout[i] / 64);
    gemm_bf16_bn_kernel<<<g, 256, 0, stream>>>(
        Ain[i], wbuf[i], (const float*)d_in[win[i] + 1], Hout[i],
        stats + i * 512, mrow[i], nout[i], kpad[i]);
    finalize_stats_kernel<<<1, 256, 0, stream>>>(
        stats + i * 512, (const float*)d_in[win[i] + 2],
        (const float*)d_in[win[i] + 3], coef + i * 512,
        nout[i], 1.f / (float)mrow[i]);
    int cnt = mrow[i] * nout[i];
    if (i == 8)
      bn_final_kernel<<<(cnt + 255) / 256, 256, 0, stream>>>(
          Hout[i], coef + i * 512, out, cnt, nout[i]);
    else
      bn_act_kernel<<<(cnt + 255) / 256, 256, 0, stream>>>(
          Hout[i], coef + i * 512, cnt, nout[i], relu[i]);
  };

  // 3) local MLP
  run_layer(0);
  run_layer(1);
  run_layer(2);
  // 4) max over K -> glb MLP
  maxk_kernel<<<(M2ROWS * 128 + 255) / 256, 256, 0, stream>>>(P3, tmax, M2ROWS, 128);
  run_layer(3);
  run_layer(4);
  // 5) concat [h | glb] -> comb MLP
  concat_kernel<<<(M1ROWS * 256 + 255) / 256, 256, 0, stream>>>(P3, tg1, P0);
  run_layer(5);
  run_layer(6);
  // 6) max over K -> cglb MLP -> f32 output
  maxk_kernel<<<(M2ROWS * 128 + 255) / 256, 256, 0, stream>>>(P2, tcmax, M2ROWS, 128);
  run_layer(7);
  run_layer(8);
}
